// FlowEmbedding_34076270526819
// MI455X (gfx1250) — matrix operand order repaired and checked by reading, not compiled, not run on hardware
//
#include <hip/hip_runtime.h>

typedef __attribute__((ext_vector_type(16))) _Float16 v16h;
typedef __attribute__((ext_vector_type(8)))  _Float16 v8h;
typedef __attribute__((ext_vector_type(2)))  _Float16 v2h;
typedef __attribute__((ext_vector_type(8)))  float    v8f;

#define BB    2
#define NN    4096
#define CIN   64
#define CO    128
#define KK    16
#define KC0   96                 /* 3 + 64 = 67 channels padded to 96 (3 x K32 chunks) */
#define MTOT  (BB * NN * KK)     /* 131072 rows */
#define MTILES (MTOT / 16)       /* 8192 */
#define GEMM_BLOCKS 1024
#define BQ_R2 1.0f
#define BN_EPS 1e-5f

// ---------------------------------------------------------------------------
// Prep: cast weights to f16 (W0 zero-padded 67 -> 96), zero BN stat buffers.
// ---------------------------------------------------------------------------
__global__ void prep_kernel(const float* __restrict__ W0,
                            const float* __restrict__ W1,
                            const float* __restrict__ W2,
                            _Float16* __restrict__ W0h,
                            _Float16* __restrict__ W1h,
                            _Float16* __restrict__ W2h,
                            float* __restrict__ stats /* 6*CO */) {
  const int tid = blockIdx.x * blockDim.x + threadIdx.x;
  const int nth = gridDim.x * blockDim.x;
  for (int e = tid; e < CO * KC0; e += nth) {
    const int o = e / KC0, c = e - o * KC0;
    W0h[e] = (c < CIN + 3) ? (_Float16)W0[o * (CIN + 3) + c] : (_Float16)0.f;
  }
  for (int e = tid; e < CO * CO; e += nth) {
    W1h[e] = (_Float16)W1[e];
    W2h[e] = (_Float16)W2[e];
  }
  for (int e = tid; e < 6 * CO; e += nth) stats[e] = 0.f;
}

// ---------------------------------------------------------------------------
// Ball query: one thread per anchor; first KK neighbor indices with d2 < r2,
// padded with the first valid index (0 if none). Early exit keeps this cheap.
// ---------------------------------------------------------------------------
__global__ void ball_query_kernel(const float* __restrict__ anchor,
                                  const float* __restrict__ neighbor,
                                  int* __restrict__ idx) {
  const int t = blockIdx.x * blockDim.x + threadIdx.x;  // b*NN + n
  if (t >= BB * NN) return;
  const int b = t / NN;
  const float ax = anchor[(size_t)t * 3 + 0];
  const float ay = anchor[(size_t)t * 3 + 1];
  const float az = anchor[(size_t)t * 3 + 2];
  const float* nb = neighbor + (size_t)b * NN * 3;
  int* out = idx + (size_t)t * KK;
  int found = 0;
  for (int j = 0; j < NN && found < KK; ++j) {
    const float dx = ax - nb[j * 3 + 0];
    const float dy = ay - nb[j * 3 + 1];
    const float dz = az - nb[j * 3 + 2];
    const float d2 = dx * dx + dy * dy + dz * dz;
    if (d2 < BQ_R2) out[found++] = j;
  }
  const int first = (found > 0) ? out[0] : 0;
  for (int k = found; k < KK; ++k) out[k] = first;
}

// ---------------------------------------------------------------------------
// Build the layer-0 activation matrix [MTOT, 96] f16:
//   c<3  : anchor - neighbor[idx]    (xyz delta)
//   3..66: features[idx] * neighbor_features[idx]
//   >=67 : zero pad
// ---------------------------------------------------------------------------
__global__ void build_feat_kernel(const float* __restrict__ anchor,
                                  const float* __restrict__ neighbor,
                                  const float* __restrict__ nfeat,
                                  const float* __restrict__ feats,
                                  const int* __restrict__ idx,
                                  _Float16* __restrict__ A0) {
  const int t = blockIdx.x;      // b*NN + n
  const int c = threadIdx.x;     // 0..95
  const int b = t / NN;
  const float av = (c < 3) ? anchor[(size_t)t * 3 + c] : 0.f;
  const int* ip = idx + (size_t)t * KK;
  for (int k = 0; k < KK; ++k) {
    const int j = ip[k];
    float v;
    if (c < 3)
      v = av - neighbor[((size_t)b * NN + j) * 3 + c];
    else if (c < 67)
      v = feats[((size_t)b * NN + j) * CIN + (c - 3)] *
          nfeat[((size_t)b * NN + j) * CIN + (c - 3)];
    else
      v = 0.f;
    A0[((size_t)t * KK + k) * KC0 + c] = (_Float16)v;
  }
}

// ---------------------------------------------------------------------------
// Fused GEMM + BN-stat layer, templated on K extent and epilogue flags.
//
// Operand assignment: WEIGHTS are the WMMA A matrix (preloaded once into
// registers); ACTIVATIONS are the B matrix, staged through a double-buffered
// LDS tile. The D tile holds (M = out-channel, N = row): each lane's 8 f32
// accumulators are 8 CONSECUTIVE channels of one row -> one packed
// global_store_b128 per lane per tile.
//
// Staging: each thread owns a CONTIGUOUS run of ELEMS halves of the tile, so
// the global load and LDS store both vectorize; the previous layer's BN
// affine + ReLU run in packed f16 (v_pk_fma_f16 / v_pk_max) with the
// per-channel scale/bias pairs precomputed into registers (tile-invariant).
//
// BN sum/sumsq accumulate per-channel in registers across all tiles; a
// 4-step shfl_xor tree + one atomicAdd pair per channel runs at the end.
// ---------------------------------------------------------------------------
template <int KC, bool AFFINE, bool RELUIN, bool RELUOUT>
__global__ __launch_bounds__(256) void gemm_bn_kernel(
    const _Float16* __restrict__ A,        /* [MTOT, KC] f16 activations */
    const _Float16* __restrict__ W,        /* [CO, KC]  f16 weights */
    _Float16* __restrict__ Out,            /* [MTOT, CO] f16 */
    float* __restrict__ sumBuf, float* __restrict__ sqBuf,
    const float* __restrict__ inScale, const float* __restrict__ inBias) {
  constexpr int NKC   = KC / 32;        // 3 (KC=96) or 4 (KC=128) WMMA chunks
  constexpr int ELEMS = 16 * KC / 256;  // contiguous halves per thread
  constexpr int PAIRS = ELEMS / 2;      // f16x2 packets per thread
  __shared__ _Float16 sA[2][16 * KC];

  const int tid  = threadIdx.x;
  const int lane = tid & 31;
  const int wave = tid >> 5;
  const int o0   = wave << 4;  // 16 output channels per wave
  const int mrow = lane & 15;
  const int hi   = lane >> 4;  // lane half selects K sub-range
  const int ncol = lane & 15;  // this lane's activation row / weight channel

  // --- Preload this wave's weight fragments (WMMA A operand) once. ---
  // A (16x32 f16) layout: lane holds channel row o0+mrow; halves 0-7 cover
  // K = 32c + 8*hi + {0..7}, halves 8-15 cover K = 32c + 16 + 8*hi + {0..7}.
  v16h wreg[NKC];
  {
    const _Float16* wrow = W + (size_t)(o0 + mrow) * KC + hi * 8;
#pragma unroll
    for (int c = 0; c < NKC; ++c) {
      const v8h w_lo = *(const v8h*)(wrow + c * 32);
      const v8h w_hi = *(const v8h*)(wrow + c * 32 + 16);
      wreg[c] = __builtin_shufflevector(w_lo, w_hi, 0, 1, 2, 3, 4, 5, 6, 7,
                                        8, 9, 10, 11, 12, 13, 14, 15);
    }
  }

  // --- Tile-invariant packed f16 affine constants for this thread's slots. ---
  v2h scp[PAIRS], bip[PAIRS];
  if (AFFINE) {
#pragma unroll
    for (int j = 0; j < PAIRS; ++j) {
      const int e = tid * ELEMS + 2 * j;
      scp[j][0] = (_Float16)inScale[e % KC];
      scp[j][1] = (_Float16)inScale[(e + 1) % KC];
      bip[j][0] = (_Float16)inBias[e % KC];
      bip[j][1] = (_Float16)inBias[(e + 1) % KC];
    }
  }

  v2h stage[PAIRS];
  auto load_tile = [&](int t) {
    const _Float16* src = A + (size_t)t * 16 * KC + tid * ELEMS;
#pragma unroll
    for (int j = 0; j < PAIRS; ++j) {
      v2h x = *(const v2h*)(src + 2 * j);
      if (AFFINE) x = x * scp[j] + bip[j];            // v_pk_fma_f16
      if (RELUIN) {
        v2h z = {};
        x = __builtin_elementwise_max(x, z);          // v_pk_max_num_f16
      }
      stage[j] = x;
    }
  };
  auto store_tile = [&](int p) {
    _Float16* dst = &sA[p][tid * ELEMS];
#pragma unroll
    for (int j = 0; j < PAIRS; ++j) *(v2h*)(dst + 2 * j) = stage[j];
  };

  float lsum[8], lsq[8];
#pragma unroll
  for (int r = 0; r < 8; ++r) lsum[r] = lsq[r] = 0.f;

  int tile = blockIdx.x;
  load_tile(tile);
  store_tile(0);
  __syncthreads();

  int p = 0;
  while (tile < MTILES) {
    const int next = tile + (int)gridDim.x;
    if (next < MTILES) {
      __builtin_prefetch(A + (size_t)next * 16 * KC, 0, 0);
      load_tile(next);  // global loads overlap the WMMAs below
    }

    // --- Activation fragments (WMMA B operand) from LDS. ---
    // B (32x16 f16) layout: lane holds column ncol (= activation row);
    // halves 0-7 cover K = 32c + 8*hi + {0..7}, halves 8-15 cover K = 32c + 16 + 8*hi + {0..7}.
    v16h act[NKC];
#pragma unroll
    for (int c = 0; c < NKC; ++c) {
      const _Float16* ap = &sA[p][ncol * KC + c * 32 + hi * 8];
      const v8h a_lo = *(const v8h*)ap;
      const v8h a_hi = *(const v8h*)(ap + 16);
      act[c] = __builtin_shufflevector(a_lo, a_hi, 0, 1, 2, 3, 4, 5, 6, 7,
                                       8, 9, 10, 11, 12, 13, 14, 15);
    }
    // Keep the ds_load clause above the WMMA chain (partial dscnt waits).
    __builtin_amdgcn_sched_barrier(0);

    v8f acc = {};
#pragma unroll
    for (int c = 0; c < NKC; ++c)
      acc = __builtin_amdgcn_wmma_f32_16x16x32_f16(false, wreg[c], false,
                                                   act[c], (short)0, acc,
                                                   false, false);

    // D layout: VGPR r holds (channel = o0 + r + 8*hi, row = m0 + ncol).
    const size_t m0 = (size_t)tile * 16;
    v8h oh;
#pragma unroll
    for (int r = 0; r < 8; ++r) {
      float v = acc[r];
      if (RELUOUT) v = fmaxf(v, 0.f);
      lsum[r] += v;
      lsq[r] += v * v;
      oh[r] = (_Float16)v;
    }
    *(v8h*)(Out + (m0 + ncol) * CO + o0 + hi * 8) = oh;  // one b128 store

    tile = next;
    if (tile < MTILES) {
      store_tile(p ^ 1);
      __syncthreads();
      p ^= 1;
    }
  }

  // --- BN statistics: reduce across the 16 rows held by this lane-half,
  // then one atomicAdd pair per channel. ---
#pragma unroll
  for (int r = 0; r < 8; ++r) {
    float s = lsum[r], q = lsq[r];
#pragma unroll
    for (int d = 1; d < 16; d <<= 1) {
      s += __shfl_xor(s, d, 32);
      q += __shfl_xor(q, d, 32);
    }
    if (ncol == 0) {
      atomicAdd(&sumBuf[o0 + hi * 8 + r], s);
      atomicAdd(&sqBuf[o0 + hi * 8 + r], q);
    }
  }
}

// ---------------------------------------------------------------------------
// Turn (sum, sumsq, gamma, beta) into per-channel affine: y = a*x + b.
// ---------------------------------------------------------------------------
__global__ void bn_finalize_kernel(const float* __restrict__ sumBuf,
                                   const float* __restrict__ sqBuf,
                                   const float* __restrict__ gamma,
                                   const float* __restrict__ beta,
                                   float* __restrict__ aOut,
                                   float* __restrict__ bOut) {
  const int c = threadIdx.x;
  if (c >= CO) return;
  const float inv = 1.f / (float)MTOT;
  const float mean = sumBuf[c] * inv;
  const float var  = sqBuf[c] * inv - mean * mean;
  const float a = gamma[c] * rsqrtf(var + BN_EPS);
  aOut[c] = a;
  bOut[c] = beta[c] - mean * a;
}

// ---------------------------------------------------------------------------
// Final: out[b,n,o] = max_k relu(a2[o]*t2 + b2[o]).
// ---------------------------------------------------------------------------
__global__ void maxpool_kernel(const _Float16* __restrict__ T2,
                               const float* __restrict__ a2,
                               const float* __restrict__ b2,
                               float* __restrict__ out) {
  const int t = blockIdx.x;   // b*NN + n
  const int o = threadIdx.x;  // channel
  const float a = a2[o], bb = b2[o];
  const _Float16* p = T2 + (size_t)t * KK * CO + o;
  float m = 0.f;  // relu floor
  for (int k = 0; k < KK; ++k) m = fmaxf(m, (float)p[k * CO] * a + bb);
  out[(size_t)t * CO + o] = m;
}

// ---------------------------------------------------------------------------
extern "C" void kernel_launch(void* const* d_in, const int* in_sizes, int n_in,
                              void* d_out, int out_size, void* d_ws,
                              size_t ws_size, hipStream_t stream) {
  (void)in_sizes; (void)n_in; (void)out_size; (void)ws_size;
  const float* anchor   = (const float*)d_in[0];
  const float* neighbor = (const float*)d_in[1];
  const float* nfeat    = (const float*)d_in[2];
  const float* feats    = (const float*)d_in[3];
  const float* W0 = (const float*)d_in[4];
  const float* g0 = (const float*)d_in[5];
  const float* be0 = (const float*)d_in[6];
  const float* W1 = (const float*)d_in[7];
  const float* g1 = (const float*)d_in[8];
  const float* be1 = (const float*)d_in[9];
  const float* W2 = (const float*)d_in[10];
  const float* g2 = (const float*)d_in[11];
  const float* be2 = (const float*)d_in[12];
  float* out = (float*)d_out;

  char* ws = (char*)d_ws;
  size_t off = 0;
  auto carve = [&](size_t bytes) -> void* {
    void* p = ws + off;
    off = (off + bytes + 255) & ~(size_t)255;
    return p;
  };

  int*      idx   = (int*)carve((size_t)MTOT * sizeof(int));
  _Float16* W0h   = (_Float16*)carve((size_t)CO * KC0 * sizeof(_Float16));
  _Float16* W1h   = (_Float16*)carve((size_t)CO * CO * sizeof(_Float16));
  _Float16* W2h   = (_Float16*)carve((size_t)CO * CO * sizeof(_Float16));
  float*    stats = (float*)carve(6 * CO * sizeof(float));
  float*    bnab  = (float*)carve(6 * CO * sizeof(float));
  _Float16* A0    = (_Float16*)carve((size_t)MTOT * KC0 * sizeof(_Float16));
  _Float16* H0    = (_Float16*)carve((size_t)MTOT * CO * sizeof(_Float16));  // s0, later t2
  _Float16* H1    = (_Float16*)carve((size_t)MTOT * CO * sizeof(_Float16));  // t1

  float* sum0 = stats + 0 * CO; float* sq0 = stats + 1 * CO;
  float* sum1 = stats + 2 * CO; float* sq1 = stats + 3 * CO;
  float* sum2 = stats + 4 * CO; float* sq2 = stats + 5 * CO;
  float* a0 = bnab + 0 * CO; float* b0 = bnab + 1 * CO;
  float* a1 = bnab + 2 * CO; float* b1 = bnab + 3 * CO;
  float* a2 = bnab + 4 * CO; float* b2 = bnab + 5 * CO;

  prep_kernel<<<64, 256, 0, stream>>>(W0, W1, W2, W0h, W1h, W2h, stats);
  ball_query_kernel<<<(BB * NN + 255) / 256, 256, 0, stream>>>(anchor, neighbor, idx);
  build_feat_kernel<<<BB * NN, KC0, 0, stream>>>(anchor, neighbor, nfeat, feats, idx, A0);

  // Layer 0: s0 = relu(A0 @ W0^T); stats over s0 (bn applied after relu).
  gemm_bn_kernel<KC0, false, false, true>
      <<<GEMM_BLOCKS, 256, 0, stream>>>(A0, W0h, H0, sum0, sq0, nullptr, nullptr);
  bn_finalize_kernel<<<1, CO, 0, stream>>>(sum0, sq0, g0, be0, a0, b0);

  // Layer 1: t1 = (a0*s0+b0) @ W1^T; stats over raw t1 (bn before relu).
  gemm_bn_kernel<CO, true, false, false>
      <<<GEMM_BLOCKS, 256, 0, stream>>>(H0, W1h, H1, sum1, sq1, a0, b0);
  bn_finalize_kernel<<<1, CO, 0, stream>>>(sum1, sq1, g1, be1, a1, b1);

  // Layer 2: t2 = relu(a1*t1+b1) @ W2^T; stats over raw t2. Writes into H0.
  gemm_bn_kernel<CO, true, true, false>
      <<<GEMM_BLOCKS, 256, 0, stream>>>(H1, W2h, H0, sum2, sq2, a1, b1);
  bn_finalize_kernel<<<1, CO, 0, stream>>>(sum2, sq2, g2, be2, a2, b2);

  // Fused bn2 + relu + max over K.
  maxpool_kernel<<<BB * NN, CO, 0, stream>>>(H0, a2, b2, out);
}
